// ONNXOutlookAttention_72353019069011
// MI455X (gfx1250) — matrix operand order repaired and checked by reading, not compiled, not run on hardware
//
#include <hip/hip_runtime.h>

// ---------------------------------------------------------------------------
// Reference collapses to:  out = blur3x3(x) @ (proj_w @ v_w)^T + proj_b
//   blur weights: w0 = e/(e+8) at offset (-1,-1), w1 = 1/(e+8) elsewhere.
// Kernel 1: fuse M = proj_w @ v_w  (256x256), store bf16 row-major in d_ws.
// Kernel 2: per (b,h) row of 64 pixels:
//   - TDM tensor_load_to_lds DMAs the full 128 KB M tile into LDS (async,
//     TENSORcnt), overlapped with the 3x3 stencil producing the bf16 A tile;
//   - 8 waves x (8 col tiles x 8 K-steps) of v_wmma_f32_16x16x32_bf16.
// ---------------------------------------------------------------------------

#define CCH 256        // channels
#define HH  64
#define WW  64
#define BB  8

typedef __attribute__((ext_vector_type(16))) __bf16 v16bf;
typedef __attribute__((ext_vector_type(8)))  float  v8f;
typedef __attribute__((ext_vector_type(4)))  unsigned int v4u;
typedef __attribute__((ext_vector_type(8)))  int v8i;
typedef __attribute__((ext_vector_type(4)))  int v4i;

union FragAB {
    uint4 u[2];
    v16bf v;
};

__device__ __forceinline__ unsigned short f2bf(float f) {
    unsigned int u = __float_as_uint(f);
    // round-to-nearest-even to bf16
    unsigned int r = (u + 0x7FFFu + ((u >> 16) & 1u)) >> 16;
    return (unsigned short)r;
}

// ---------------------------------------------------------------------------
// Kernel 1: M[e, c] = sum_d proj_w[e, d] * v_w[d, c]   (bf16 output in ws)
// ---------------------------------------------------------------------------
__global__ __launch_bounds__(256)
void fuse_weights_kernel(const float* __restrict__ v_w,
                         const float* __restrict__ proj_w,
                         unsigned short* __restrict__ Mbf) {
    const int e = blockIdx.x;
    const int c = threadIdx.x;
    float acc = 0.0f;
#pragma unroll 8
    for (int d = 0; d < CCH; ++d) {
        acc += proj_w[e * CCH + d] * v_w[d * CCH + c];
    }
    Mbf[e * CCH + c] = f2bf(acc);
}

// ---------------------------------------------------------------------------
// Kernel 2: fused stencil + GEMM.
// LDS: sB = M, e-major, 256*256 bf16 = 128 KB   (TDM destination)
//      sA = blurred pixels, pixel-major, 64*256 bf16 = 32 KB
// ---------------------------------------------------------------------------
__global__ __launch_bounds__(256)
void outlook_fused_kernel(const float* __restrict__ x,
                          const unsigned short* __restrict__ Mbf,
                          const float* __restrict__ proj_b,
                          float* __restrict__ out) {
    extern __shared__ char smem[];
    unsigned short* sB = (unsigned short*)smem;                   // [e*256 + c]
    unsigned short* sA = (unsigned short*)(smem + CCH * CCH * 2); // [pix*256 + c]

    const int t    = threadIdx.x;
    const int bh   = blockIdx.x;       // 0 .. B*H-1
    const int b    = bh >> 6;          // H == 64
    const int h    = bh & 63;
    const long pixBase = (long)bh * WW;

    // ---- Phase 1: DMA full B matrix (M, bf16) into LDS via the TDM -------
#if __has_builtin(__builtin_amdgcn_tensor_load_to_lds)
    if (t < 32) {   // one wave issues the descriptor (EXEC ignored by TDM)
        const unsigned long long ga = (unsigned long long)(uintptr_t)Mbf;
        const unsigned ldsOff = (unsigned)(uintptr_t)sB; // LDS byte offset

        // D# group 0: count=1 | lds_addr | global_addr[56:0] | type=2
        v4u g0;
        g0.x = 1u;
        g0.y = ldsOff;
        g0.z = (unsigned)(ga & 0xFFFFFFFFu);
        g0.w = (unsigned)((ga >> 32) & 0x01FFFFFFu) | (2u << 30);

        // D# group 1: data_size=1 (2B) | tensor_dim0/1=256 | tile_dim0/1=256
        //             | tensor_dim0_stride=256 ; no multicast / barrier / pad
        v8i g1;
        g1[0] = (int)(1u << 16);                 // data_size = 2 bytes
        g1[1] = (int)((256u & 0xFFFFu) << 16);   // tensor_dim0 lo16 @ bits 63:48
        g1[2] = (int)((256u & 0xFFFFu) << 16);   // td0 hi16=0 | tensor_dim1 lo16
        g1[3] = (int)(256u << 16);               // td1 hi16=0 | tile_dim0 = 256
        g1[4] = (int)(256u);                     // tile_dim1 = 256, tile_dim2 = 0
        g1[5] = (int)(256u);                     // tensor_dim0_stride lo32
        g1[6] = 0;                               // stride0 hi | stride1 lo
        g1[7] = 0;

        v4i gz4 = {0, 0, 0, 0};                  // groups 2/3: unused (2D tensor)
        v8i gz8 = {0, 0, 0, 0, 0, 0, 0, 0};      // trailing group: unused
        __builtin_amdgcn_tensor_load_to_lds(g0, g1, gz4, gz4, gz8, 0);
    }
#else
    {
        const uint4* g = (const uint4*)Mbf;      // 8192 x 16B synchronous copy
        uint4* s = (uint4*)sB;
#pragma unroll
        for (int i = 0; i < 32; ++i) {
            int idx = t + i * 256;
            s[idx] = g[idx];
        }
    }
#endif

    // ---- Phase 2 (overlapped with TDM): 3x3 blur -> bf16 A tile ----------
    {
        const float kE = 2.71828182845904523536f;
        const float W1 = 1.0f / (kE + 8.0f);
        const float W0 = kE / (kE + 8.0f);

        const int c0 = (t & 63) << 2;   // 4 consecutive channels
        const int wq = t >> 6;          // 0..3
#pragma unroll 4
        for (int it = 0; it < 16; ++it) {
            const int w = wq * 16 + it;
            float ax = 0.f, ay = 0.f, az = 0.f, aw = 0.f;
#pragma unroll
            for (int dy = -1; dy <= 1; ++dy) {
                const int hh = h + dy;
                const bool vy = (hh >= 0) && (hh < HH);
#pragma unroll
                for (int dx = -1; dx <= 1; ++dx) {
                    const int ww2 = w + dx;
                    if (vy && ww2 >= 0 && ww2 < WW) {
                        const float wt = (dy == -1 && dx == -1) ? W0 : W1;
                        const float4 p = *(const float4*)(x +
                            ((((long)b * HH + hh) * WW + ww2) * CCH + c0));
                        ax += wt * p.x; ay += wt * p.y;
                        az += wt * p.z; aw += wt * p.w;
                    }
                }
            }
            uint2 pk;
            pk.x = (unsigned int)f2bf(ax) | ((unsigned int)f2bf(ay) << 16);
            pk.y = (unsigned int)f2bf(az) | ((unsigned int)f2bf(aw) << 16);
            *(uint2*)&sA[w * CCH + c0] = pk;
        }
    }

#if __has_builtin(__builtin_amdgcn_tensor_load_to_lds)
    __builtin_amdgcn_s_wait_tensorcnt(0);   // no-op for waves with count 0
#endif
    __syncthreads();

    // ---- Phase 3: WMMA GEMM  out[64 x 256] = A[64 x 256] @ B^T ----------
    const int lane    = t & 31;
    const int wave    = t >> 5;
    const int rowTile = wave & 3;        // 16-pixel row tile
    const int colHalf = wave >> 2;       // col tiles [colHalf*8, +8)
    const int n       = lane & 15;       // column-within-tile owned by lane
    const int kh      = lane >> 4;       // lane half selects K sub-range

    const int pixL = rowTile * 16 + n;   // A row (pixel) this lane feeds

    v8f acc[8];
#pragma unroll
    for (int ct = 0; ct < 8; ++ct) {
        v8f z = {0.f, 0.f, 0.f, 0.f, 0.f, 0.f, 0.f, 0.f};
        acc[ct] = z;
    }

#pragma unroll
    for (int k0 = 0; k0 < CCH; k0 += 32) {
        // A fragment (16x32 bf16): lanes 0-15 K = k0+{0..7,16..23},
        //                          lanes 16-31 K = k0+{8..15,24..31}
        FragAB fa;
        fa.u[0] = *(const uint4*)&sA[pixL * CCH + k0 + kh * 8];
        fa.u[1] = *(const uint4*)&sA[pixL * CCH + k0 + 16 + kh * 8];

#pragma unroll
        for (int ct = 0; ct < 8; ++ct) {
            const int e = (colHalf * 8 + ct) * 16 + n;    // B column (out chan)
            FragAB fb;   // 32x16 bf16: lane holds K = k0 + kh*8 + {0..7,16..23}
            fb.u[0] = *(const uint4*)&sB[e * CCH + k0 + kh * 8];
            fb.u[1] = *(const uint4*)&sB[e * CCH + k0 + 16 + kh * 8];

            acc[ct] = __builtin_amdgcn_wmma_f32_16x16x32_bf16(
                false, fa.v, false, fb.v, (short)0, acc[ct], false, false);
        }
    }

    // ---- Phase 4: bias + store ------------------------------------------
#pragma unroll
    for (int ct = 0; ct < 8; ++ct) {
        const int e = (colHalf * 8 + ct) * 16 + n;
        const float bias = proj_b[e];
#pragma unroll
        for (int j = 0; j < 8; ++j) {
            const int m = j + kh * 8;                 // pixel row in D layout
            const long pix = pixBase + rowTile * 16 + m;
            out[pix * CCH + e] = acc[ct][j] + bias;
        }
    }
}

// ---------------------------------------------------------------------------
extern "C" void kernel_launch(void* const* d_in, const int* in_sizes, int n_in,
                              void* d_out, int out_size, void* d_ws, size_t ws_size,
                              hipStream_t stream) {
    const float* x      = (const float*)d_in[0];
    const float* v_w    = (const float*)d_in[1];
    const float* proj_w = (const float*)d_in[2];
    const float* proj_b = (const float*)d_in[3];
    float* out = (float*)d_out;

    unsigned short* Mbf = (unsigned short*)d_ws;   // 256*256*2 = 128 KB scratch

    // Kernel 1: fuse the two linear layers: M = proj_w @ v_w (bf16)
    fuse_weights_kernel<<<CCH, 256, 0, stream>>>(v_w, proj_w, Mbf);

    // Kernel 2: fused stencil + WMMA GEMM. LDS: 128 KB (B) + 32 KB (A).
    const size_t shmem = (size_t)CCH * CCH * 2 + (size_t)WW * CCH * 2; // 160 KB
    outlook_fused_kernel<<<BB * HH, 256, shmem, stream>>>(x, Mbf, proj_b, out);
}
